// HeteroGATReasoner_22746146799799
// MI455X (gfx1250) — matrix operand order repaired and checked by reading, not compiled, not run on hardware
//
#include <hip/hip_runtime.h>

// ---------------------------------------------------------------------------
// HeteroGATv2 reasoner for MI455X (gfx1250, wave32, WMMA).
// GEMMs use v_wmma_f32_16x16x32_f16 (f16 in / f32 accumulate).
// Edge softmax / aggregation uses wave-per-edge VALU + f32 global atomics.
// ---------------------------------------------------------------------------

typedef __attribute__((ext_vector_type(16))) _Float16 v16h;
typedef __attribute__((ext_vector_type(8)))  _Float16 v8h;
typedef __attribute__((ext_vector_type(8)))  float    v8f;

// ---------------- atomic helpers ----------------
__device__ __forceinline__ void atomicAddF32(float* addr, float v) {
    __hip_atomic_fetch_add(addr, v, __ATOMIC_RELAXED, __HIP_MEMORY_SCOPE_AGENT);
}

// float max via monotone integer-bits trick (correct for mixed signs)
__device__ __forceinline__ void atomicMaxF32(float* addr, float val) {
    if (val >= 0.0f)
        atomicMax((int*)addr, __float_as_int(val));
    else
        atomicMin((unsigned int*)addr, (unsigned int)__float_as_int(val));
}

// ---------------- WMMA GEMM:  C[M,N] = A[M,K](f16) * B[K,N] + bias ----------
// B is supplied pre-transposed as BT[N,K] (f16) so per-lane fragment loads are
// contiguous.  One wave computes a 16x64 tile; A fragment reused for 4 WMMAs.
// Requires: M%16==0, N%64==0, K%32==0 (true for all call sites here).
__global__ __launch_bounds__(256)
void wmma_gemm_bias(const _Float16* __restrict__ A,
                    const _Float16* __restrict__ BT,
                    const float* __restrict__ bias,
                    float* __restrict__ C,
                    int M, int N, int K) {
    const int lane = threadIdx.x & 31;
    const int gw   = (blockIdx.x * blockDim.x + threadIdx.x) >> 5;
    const int ntiles = N >> 6;                 // 64-wide N tiles
    const int mt = gw / ntiles;
    const int nt = gw % ntiles;
    if (mt * 16 >= M) return;                  // wave-uniform exit (EXEC all-1 inside)

    const int arow = lane & 15;
    const int ksel = lane >> 4;                // 0 or 1 (half-wave select)
    const _Float16* __restrict__ arow_ptr = A + (size_t)(mt * 16 + arow) * K;

    v8f acc[4] = {};
    for (int kk = 0; kk < K; kk += 32) {
        // A fragment (16x32 f16): lanes 0-15 K=[0..15], lanes 16-31 K=[16..31]
        v16h afrag = *(const v16h*)(arow_ptr + kk + ksel * 16);
#pragma unroll
        for (int t = 0; t < 4; ++t) {
            // B fragment (32x16 f16): lane = column, K contiguous in BT[N][K]
            const int col = nt * 64 + t * 16 + (lane & 15);
            v16h bfrag = *(const v16h*)(BT + (size_t)col * K + kk + ksel * 16);
            acc[t] = __builtin_amdgcn_wmma_f32_16x16x32_f16(
                         false, afrag, false, bfrag, (short)0, acc[t], false, false);
        }
    }
    // D layout: VGPR r -> M = r (lanes 0-15) / r+8 (lanes 16-31), N = lane&15
    const int rbase = mt * 16 + ksel * 8;
#pragma unroll
    for (int t = 0; t < 4; ++t) {
        const int col = nt * 64 + t * 16 + (lane & 15);
        const float b = bias ? bias[col] : 0.0f;
#pragma unroll
        for (int r = 0; r < 8; ++r)
            C[(size_t)(rbase + r) * N + col] = acc[t][r] + b;
    }
}

// ---------------- small utility kernels ----------------
__global__ void fill_f32(float* __restrict__ p, float v, long long n) {
    long long i = (long long)blockIdx.x * blockDim.x + threadIdx.x;
    if (i < n) p[i] = v;
}

__global__ void cvt_f16(const float* __restrict__ in, _Float16* __restrict__ out, long long n) {
    long long i = (long long)blockIdx.x * blockDim.x + threadIdx.x;
    if (i < n) out[i] = (_Float16)in[i];
}

__global__ void relu_cvt(const float* __restrict__ in, float* __restrict__ of,
                         _Float16* __restrict__ oh, long long n) {
    long long i = (long long)blockIdx.x * blockDim.x + threadIdx.x;
    if (i >= n) return;
    float v = in[i];
    v = v > 0.0f ? v : 0.0f;
    of[i] = v;
    oh[i] = (_Float16)v;
}

// WT[n*K + k] = (f16) W[k*N + n]   (row-major [K,N] -> transposed f16 [N,K])
__global__ void transpose_to_f16(const float* __restrict__ W, _Float16* __restrict__ WT,
                                 int K, int N) {
    int i = blockIdx.x * blockDim.x + threadIdx.x;
    if (i >= K * N) return;
    int n = i / K, k = i % K;
    WT[(size_t)n * K + k] = (_Float16)W[(size_t)k * N + n];
}

__global__ void add_bias_rows(float* __restrict__ out, const float* __restrict__ bias,
                              int Nn, int C) {
    int i = blockIdx.x * blockDim.x + threadIdx.x;
    if (i >= Nn * C) return;
    out[i] += bias[i % C];
}

// tiny-K projection (Creator K=2, Licensee K=1)
__global__ void proj_small(const float* __restrict__ x, const float* __restrict__ W,
                           const float* __restrict__ b, float* __restrict__ h,
                           int Nn, int K, int C) {
    int i = blockIdx.x * blockDim.x + threadIdx.x;
    if (i >= Nn * C) return;
    int n = i / C, c = i % C;
    float acc = b[c];
    for (int k = 0; k < K; ++k) acc += x[(size_t)n * K + k] * W[(size_t)k * C + c];
    h[i] = acc;
}

// ---------------- GATv2 edge kernels ----------------
// pass 1: wave per (edge, head): logit = att . leakyrelu(xl[s]+xr[d]+ea*We)
__global__ void edge_logit(const float* __restrict__ xl, const float* __restrict__ xr,
                           const float* __restrict__ ea, const int* __restrict__ src,
                           const int* __restrict__ dst, const float* __restrict__ We,
                           const float* __restrict__ att, float* __restrict__ logit,
                           float* __restrict__ m, int E, int H, int C) {
    int w = (blockIdx.x * blockDim.x + threadIdx.x) >> 5;
    int lane = threadIdx.x & 31;
    if (w >= E * H) return;
    int e = w / H, h = w % H;
    int s = src[e], d = dst[e];
    float eav = ea[e];
    const float* pl = xl + ((size_t)s * H + h) * C;
    const float* pr = xr + ((size_t)d * H + h) * C;
    const float* pw = We + h * C;
    const float* pa = att + h * C;
    float acc = 0.0f;
    for (int c = lane; c < C; c += 32) {
        float v = pl[c] + pr[c] + eav * pw[c];
        v = v > 0.0f ? v : 0.2f * v;                 // leaky_relu(., 0.2)
        acc = fmaf(pa[c], v, acc);
    }
#pragma unroll
    for (int off = 16; off; off >>= 1) acc += __shfl_xor(acc, off, 32);
    if (lane == 0) {
        logit[w] = acc;
        atomicMaxF32(&m[(size_t)d * H + h], acc);
    }
}

// pass 2: ex = exp(logit - m[dst]); denom[dst] += ex
__global__ void edge_exp(const float* __restrict__ logit, const int* __restrict__ dst,
                         const float* __restrict__ m, float* __restrict__ ex,
                         float* __restrict__ denom, int E, int H) {
    int i = blockIdx.x * blockDim.x + threadIdx.x;
    if (i >= E * H) return;
    int e = i / H, h = i % H;
    int d = dst[e];
    float v = __expf(logit[i] - m[(size_t)d * H + h]);
    ex[i] = v;
    atomicAddF32(&denom[(size_t)d * H + h], v);
}

// pass 3: out[dst,c] += mean_h( alpha_h * xl[src,h,c] )
__global__ void edge_scatter(const float* __restrict__ xl, const float* __restrict__ ex,
                             const float* __restrict__ denom, const int* __restrict__ src,
                             const int* __restrict__ dst, float* __restrict__ out,
                             int E, int H, int C) {
    long long i = (long long)blockIdx.x * blockDim.x + threadIdx.x;
    if (i >= (long long)E * C) return;
    int e = (int)(i / C), c = (int)(i % C);
    int s = src[e], d = dst[e];
    float v = 0.0f;
    for (int h = 0; h < H; ++h) {
        float a = ex[(size_t)e * H + h] / (denom[(size_t)d * H + h] + 1e-16f);
        v += a * xl[((size_t)s * H + h) * C + c];
    }
    atomicAddF32(&out[(size_t)d * C + c], v / (float)H);
}

// ---------------- classifier head (single asset row) ----------------
__global__ void classifier(const float* __restrict__ h_a, const int* __restrict__ qidx,
                           const float* __restrict__ Wc1, const float* __restrict__ bc1,
                           const float* __restrict__ Wc2, const float* __restrict__ bc2,
                           float* __restrict__ out) {
    __shared__ float hid[128];
    const int q = qidx[0];
    const float* a = h_a + (size_t)q * 128;
    int c = threadIdx.x;
    float acc = bc1[c];
    for (int k = 0; k < 128; ++k) acc += a[k] * Wc1[(size_t)k * 128 + c];
    hid[c] = acc > 0.0f ? acc : 0.0f;
    __syncthreads();
    if (c < 3) {
        float o = bc2[c];
        for (int k = 0; k < 128; ++k) o += hid[k] * Wc2[(size_t)k * 3 + c];
        out[c] = o;
    }
}

// ---------------------------------------------------------------------------
extern "C" void kernel_launch(void* const* d_in, const int* in_sizes, int n_in,
                              void* d_out, int out_size, void* d_ws, size_t ws_size,
                              hipStream_t stream) {
    (void)in_sizes; (void)n_in; (void)out_size; (void)ws_size;
    constexpr int NA = 50000, NC = 10000, NL = 10000, C = 128;
    const int counts[3] = {NA, NC, NL};

    // ---- inputs (setup_inputs dict order) ----
    const float* x_a = (const float*)d_in[0];
    const float* x_c = (const float*)d_in[1];
    const float* x_l = (const float*)d_in[2];
    const int*   e_src[6]; const int* e_dst[6]; const float* e_ea[6];
    const int    Ecnt[6]   = {50000, 100000, 200000, 100000, 50000, 100000};
    const int    relSrc[6] = {0, 0, 0, 0, 1, 2};   // 0=Asset 1=Creator 2=Licensee
    const int    relDst[6] = {1, 2, 0, 0, 0, 0};
    for (int i = 0; i < 6; ++i) {
        e_src[i] = (const int*)d_in[3 + 3 * i];
        e_dst[i] = (const int*)d_in[4 + 3 * i];
        e_ea[i]  = (const float*)d_in[5 + 3 * i];
    }
    const float* Wp_a = (const float*)d_in[21]; const float* bp_a = (const float*)d_in[22];
    const float* Wp_c = (const float*)d_in[23]; const float* bp_c = (const float*)d_in[24];
    const float* Wp_l = (const float*)d_in[25]; const float* bp_l = (const float*)d_in[26];
    const float* Wl1 = (const float*)d_in[27]; const float* bl1 = (const float*)d_in[28];
    const float* Wr1 = (const float*)d_in[29]; const float* br1 = (const float*)d_in[30];
    const float* We1 = (const float*)d_in[31]; const float* att1 = (const float*)d_in[32];
    const float* bias1 = (const float*)d_in[33];
    const float* Wl2 = (const float*)d_in[34]; const float* bl2 = (const float*)d_in[35];
    const float* Wr2 = (const float*)d_in[36]; const float* br2 = (const float*)d_in[37];
    const float* We2 = (const float*)d_in[38]; const float* att2 = (const float*)d_in[39];
    const float* bias2 = (const float*)d_in[40];
    const float* Wc1 = (const float*)d_in[41]; const float* bc1 = (const float*)d_in[42];
    const float* Wc2 = (const float*)d_in[43]; const float* bc2 = (const float*)d_in[44];
    const int*   qidx = (const int*)d_in[45];

    // ---- workspace carve-out ----
    char* wp = (char*)d_ws;
    auto carve = [&](size_t bytes) -> char* {
        char* p = wp;
        wp += (bytes + 255) & ~(size_t)255;
        return p;
    };
    float* h[3];  float* o[3];  _Float16* h16[3];
    h[0] = (float*)carve((size_t)NA * C * 4);
    h[1] = (float*)carve((size_t)NC * C * 4);
    h[2] = (float*)carve((size_t)NL * C * 4);
    o[0] = (float*)carve((size_t)NA * C * 4);
    o[1] = (float*)carve((size_t)NC * C * 4);
    o[2] = (float*)carve((size_t)NL * C * 4);
    h16[0] = (_Float16*)carve((size_t)NA * C * 2);
    h16[1] = (_Float16*)carve((size_t)NC * C * 2);
    h16[2] = (_Float16*)carve((size_t)NL * C * 2);
    _Float16* xa16  = (_Float16*)carve((size_t)NA * 512 * 2);
    float*    xlbuf = (float*)carve((size_t)NA * 256 * 4);
    float*    xrbuf = (float*)carve((size_t)NA * 256 * 4);
    float*    lbuf  = (float*)carve((size_t)200000 * 2 * 4);
    float*    exbuf = (float*)carve((size_t)200000 * 2 * 4);
    float*    mbuf  = (float*)carve((size_t)NA * 2 * 4);
    float*    dbuf  = (float*)carve((size_t)NA * 2 * 4);
    _Float16* wT1   = (_Float16*)carve((size_t)512 * 128 * 2);
    _Float16* wT2   = (_Float16*)carve((size_t)512 * 128 * 2);

    auto nblk = [](long long n, int t) { return (unsigned)((n + t - 1) / t); };
    auto fill = [&](float* p, float v, long long n) {
        fill_f32<<<nblk(n, 256), 256, 0, stream>>>(p, v, n);
    };
    auto gemm = [&](const _Float16* A, const _Float16* BT, const float* bias,
                    float* Cout, int M, int N, int K) {
        long long waves = (long long)(M / 16) * (N / 64);
        wmma_gemm_bias<<<nblk(waves, 8), 256, 0, stream>>>(A, BT, bias, Cout, M, N, K);
    };

    // ---- stage 1: projections ----
    cvt_f16<<<nblk((long long)NA * 512, 256), 256, 0, stream>>>(x_a, xa16, (long long)NA * 512);
    transpose_to_f16<<<nblk(512 * 128, 256), 256, 0, stream>>>(Wp_a, wT1, 512, C);
    gemm(xa16, wT1, bp_a, h[0], NA, C, 512);
    proj_small<<<nblk((long long)NC * C, 256), 256, 0, stream>>>(x_c, Wp_c, bp_c, h[1], NC, 2, C);
    proj_small<<<nblk((long long)NL * C, 256), 256, 0, stream>>>(x_l, Wp_l, bp_l, h[2], NL, 1, C);
    for (int t = 0; t < 3; ++t)
        cvt_f16<<<nblk((long long)counts[t] * C, 256), 256, 0, stream>>>(
            h[t], h16[t], (long long)counts[t] * C);

    // ---- hetero GATv2 layer ----
    auto run_layer = [&](int H, const float* Wl, const float* bl, const float* Wr,
                         const float* br, const float* We, const float* att,
                         const float* bias) {
        const int HC = H * C;
        for (int t = 0; t < 3; ++t) fill(o[t], 0.0f, (long long)counts[t] * C);
        for (int i = 0; i < 6; ++i) {
            const int st = relSrc[i], dt = relDst[i], E = Ecnt[i];
            const int Ns = counts[st], Nd = counts[dt];
            // per-relation weights -> transposed f16
            transpose_to_f16<<<nblk(C * HC, 256), 256, 0, stream>>>(
                Wl + (size_t)i * C * HC, wT1, C, HC);
            transpose_to_f16<<<nblk(C * HC, 256), 256, 0, stream>>>(
                Wr + (size_t)i * C * HC, wT2, C, HC);
            // xl = h_src @ Wl + bl ; xr = h_dst @ Wr + br     (WMMA)
            gemm(h16[st], wT1, bl + (size_t)i * HC, xlbuf, Ns, HC, C);
            gemm(h16[dt], wT2, br + (size_t)i * HC, xrbuf, Nd, HC, C);
            // segment softmax over incoming edges
            fill(mbuf, -3.0e38f, (long long)Nd * H);
            fill(dbuf, 0.0f, (long long)Nd * H);
            edge_logit<<<nblk((long long)E * H * 32, 256), 256, 0, stream>>>(
                xlbuf, xrbuf, e_ea[i], e_src[i], e_dst[i],
                We + (size_t)i * HC, att + (size_t)i * HC, lbuf, mbuf, E, H, C);
            edge_exp<<<nblk((long long)E * H, 256), 256, 0, stream>>>(
                lbuf, e_dst[i], mbuf, exbuf, dbuf, E, H);
            edge_scatter<<<nblk((long long)E * C, 256), 256, 0, stream>>>(
                xlbuf, exbuf, dbuf, e_src[i], e_dst[i], o[dt], E, H, C);
            // per-relation bias broadcast to ALL dst nodes (reference semantics)
            add_bias_rows<<<nblk((long long)Nd * C, 256), 256, 0, stream>>>(
                o[dt], bias + (size_t)i * C, Nd, C);
        }
        for (int t = 0; t < 3; ++t)
            relu_cvt<<<nblk((long long)counts[t] * C, 256), 256, 0, stream>>>(
                o[t], h[t], h16[t], (long long)counts[t] * C);
    };

    run_layer(2, Wl1, bl1, Wr1, br1, We1, att1, bias1);   // heads=2
    run_layer(1, Wl2, bl2, Wr2, br2, We2, att2, bias2);   // heads=1

    // ---- classifier head on queried asset ----
    classifier<<<1, 128, 0, stream>>>(h[0], qidx, Wc1, bc1, Wc2, bc2, (float*)d_out);
}
